// CodebookEMA_55293408969397
// MI455X (gfx1250) — matrix operand order repaired and checked by reading, not compiled, not run on hardware
//
#include <hip/hip_runtime.h>

#define NVEC   32768   // B*H*W = 32*32*32
#define KCODE  1024
#define DIM    256
#define HW     1024    // H*W
#define BETA   0.25f
#define DECAY  0.99f
#define EPSV   1e-5f

typedef __attribute__((ext_vector_type(16))) __bf16        v16bf;
typedef __attribute__((ext_vector_type(8)))  float         v8f;
typedef __attribute__((ext_vector_type(4)))  unsigned int  v4u;
typedef __attribute__((ext_vector_type(8)))  int           v8i;
typedef __attribute__((ext_vector_type(4)))  int           v4i;

#if defined(__gfx1250__) && __has_builtin(__builtin_amdgcn_tensor_load_to_lds) && \
    __has_builtin(__builtin_amdgcn_s_wait_tensorcnt)
#define USE_TDM 1
#else
#define USE_TDM 0
#endif

__device__ __forceinline__ __bf16 f2bf(float f) {
  // round-to-nearest-even fp32 -> bf16
  unsigned u = __builtin_bit_cast(unsigned, f);
  unsigned r = u + 0x7fffu + ((u >> 16) & 1u);
  return __builtin_bit_cast(__bf16, (unsigned short)(r >> 16));
}

#if USE_TDM
// Contiguous 8KB (16 codes x 256 dims bf16) DMA: global -> LDS via the
// Tensor Data Mover.  1-D copy: tile_dim0 = tensor_dim0 = 4096 elems of 2B.
// This toolchain exposes the 6-arg builtin:
//   (uint32x4 g0, int32x8 g1, int32x4 g2, int32x4 g3, int32x8 pad, i32 cpol)
__device__ __forceinline__ void tdm_tile_to_lds(const __bf16* gsrc, unsigned lds_off) {
  unsigned long long ga = (unsigned long long)gsrc;
  v4u g0;
  g0[0] = 1u;                                                  // count=1 (valid user D#)
  g0[1] = lds_off;                                             // lds_addr (bytes)
  g0[2] = (unsigned)ga;                                        // global_addr[31:0]
  g0[3] = ((unsigned)(ga >> 32) & 0x01FFFFFFu) | (2u << 30);   // addr[56:32] | type=2
  v8i g1;
  g1[0] = (1 << 16);                  // workgroup_mask=0, data_size=1 (2 bytes)
  g1[1] = (int)(4096u << 16);         // tensor_dim0[15:0]=4096 (bits 63:48)
  g1[2] = (1 << 16);                  // tensor_dim0 hi=0; tensor_dim1 lo=1
  g1[3] = (int)(4096u << 16);         // tensor_dim1 hi=0; tile_dim0=4096
  g1[4] = 1;                          // tile_dim1=1, tile_dim2=0
  g1[5] = 4096;                       // tensor_dim0_stride lo32
  g1[6] = 0;                          // stride hi / dim1_stride lo
  g1[7] = 0;                          // dim1_stride hi
  v4i zz = {0, 0, 0, 0};              // groups 2/3 unused (<=2D tensor)
  v8i z8 = {0, 0, 0, 0, 0, 0, 0, 0};
  __builtin_amdgcn_tensor_load_to_lds(g0, g1, zz, zz, z8, 0);
}
#endif

// ---------------------------------------------------------------------------
// Kernel 0: codebook -> bf16 copy (L2-resident operand for WMMA) + ||e||^2
// ---------------------------------------------------------------------------
__global__ void __launch_bounds__(DIM)
prep_codebook(const float* __restrict__ emb, __bf16* __restrict__ ebf,
              float* __restrict__ enorm) {
  const int k = blockIdx.x, d = threadIdx.x;
  float v = emb[k * DIM + d];
  ebf[k * DIM + d] = f2bf(v);
  __shared__ float red[DIM];
  red[d] = v * v;
  __syncthreads();
  for (int s = DIM / 2; s > 0; s >>= 1) {
    if (d < s) red[d] += red[d + s];
    __syncthreads();
  }
  if (d == 0) enorm[k] = red[0];
}

// ---------------------------------------------------------------------------
// Kernel 1: zero atomic accumulators (counts, dw, loss) every call
// ---------------------------------------------------------------------------
__global__ void zero_ws(float* __restrict__ counts, float* __restrict__ dw,
                        float* __restrict__ loss) {
  int i = blockIdx.x * blockDim.x + threadIdx.x;
  if (i < KCODE) counts[i] = 0.0f;
  if (i < KCODE * DIM) dw[i] = 0.0f;
  if (i == 0) *loss = 0.0f;
}

// ---------------------------------------------------------------------------
// Kernel 2: fused distance GEMM (bf16 WMMA, f32 accum) + argmin.
// Each wave owns 16 z-rows (A panel in registers for the whole sweep).
// Codebook tiles (16 codes x 256 dims, 8KB) are staged in LDS, shared by all
// 8 waves, double-buffered, fetched by the TDM (wave 0 issues; everyone
// barriers) with s_wait_tensorcnt for completion.
// ---------------------------------------------------------------------------
__global__ void __launch_bounds__(256)
argmin_kernel(const float* __restrict__ z, const __bf16* __restrict__ ebf,
              const float* __restrict__ enorm, int* __restrict__ idx_i,
              float* __restrict__ idx_f) {
  __shared__ __attribute__((aligned(32))) __bf16 es[2][16 * DIM];  // 2 x 8KB
  __shared__ float ens[KCODE];                                     // 4KB

  const int tid  = threadIdx.x;
  const int lane = tid & 31;
  const int wave = tid >> 5;
  const int half = lane >> 4;      // which 16-lane half of the wave
  const int nl   = lane & 15;
  const int rowbase = blockIdx.x * 128 + wave * 16;

  // This lane supplies A-matrix row m = rowbase + nl (z vector index n).
  const int n  = rowbase + nl;
  const int b  = n >> 10;          // batch
  const int hw = n & (HW - 1);     // h*32+w
  const float* zb = z + (size_t)b * DIM * HW + hw;  // + c*HW walks channels

  // A fragments: 8 chunks of K=32, contiguous K order (matches B):
  //   slots 0..7  -> K = kb + 16*half + e
  //   slots 8..15 -> K = kb + 16*half + 8 + (e-8)
  v16bf afrag[8];
#pragma unroll
  for (int c = 0; c < 8; ++c) {
    const int kb = c * 32;
#pragma unroll
    for (int e = 0; e < 8; ++e) {
      afrag[c][e]     = f2bf(zb[(size_t)(kb + 16 * half + e) * HW]);
      afrag[c][8 + e] = f2bf(zb[(size_t)(kb + 16 * half + 8 + e) * HW]);
    }
  }

  // Stage ||e||^2 into LDS (shared by all tiles).
  for (int i = tid; i < KCODE; i += 256) ens[i] = enorm[i];

  // Prologue: fetch tile 0 into buffer 0.
#if USE_TDM
  if (wave == 0) {
    tdm_tile_to_lds(ebf, (unsigned)(__SIZE_TYPE__)&es[0][0]);
    __builtin_amdgcn_s_wait_tensorcnt(0);
  }
#else
  {
    const v4u* src = (const v4u*)ebf;
    v4u* dst = (v4u*)&es[0][0];
    dst[tid] = src[tid];
    dst[tid + 256] = src[tid + 256];
  }
#endif
  __syncthreads();

  float bestv[8];
  int   besti[8];
#pragma unroll
  for (int r = 0; r < 8; ++r) { bestv[r] = 3.4e38f; besti[r] = 0; }

  for (int t = 0; t < KCODE / 16; ++t) {
    const int buf = t & 1;

    // Prefetch next tile into the other buffer (overlaps with WMMA below).
#if USE_TDM
    if (wave == 0 && t + 1 < KCODE / 16)
      tdm_tile_to_lds(ebf + (size_t)(t + 1) * 16 * DIM,
                      (unsigned)(__SIZE_TYPE__)&es[buf ^ 1][0]);
#else
    if (t + 1 < KCODE / 16) {
      const v4u* src = (const v4u*)(ebf + (size_t)(t + 1) * 16 * DIM);
      v4u* dst = (v4u*)&es[buf ^ 1][0];
      dst[tid] = src[tid];
      dst[tid + 256] = src[tid + 256];
    }
#endif

    const int nb   = t * 16;
    const int code = nb + nl;                 // this lane's D-matrix column
    const float en = ens[code];
    // B fragment: lane = column n, slot j -> K = c*32 + 16*half + j
    // -> 32 contiguous bytes of LDS codebook row `nl`
    const __bf16* erow = &es[buf][nl * DIM + 16 * half];
    v8f acc = {};
#pragma unroll
    for (int c = 0; c < 8; ++c) {
      v16bf bfrag = *(const v16bf*)(erow + c * 32);
      acc = __builtin_amdgcn_wmma_f32_16x16x32_bf16(
          false, afrag[c], false, bfrag, (short)0, acc, false, false);
    }
#pragma unroll
    for (int r = 0; r < 8; ++r) {
      // D layout: lane l holds column nl, VGPR r holds row M = r + 8*half
      float d = en - 2.0f * acc[r];
      if (d < bestv[r]) { bestv[r] = d; besti[r] = code; }
    }

#if USE_TDM
    if (wave == 0) __builtin_amdgcn_s_wait_tensorcnt(0);
#endif
    __syncthreads();
  }

  // Cross-lane argmin over the 16 lanes of each half (xor masks < 16 stay
  // inside the half). Tie-break toward the smaller code index.
#pragma unroll
  for (int r = 0; r < 8; ++r) {
    float v = bestv[r];
    int   i = besti[r];
#pragma unroll
    for (int m = 1; m < 16; m <<= 1) {
      float ov = __shfl_xor(v, m, 32);
      int   oi = __shfl_xor(i, m, 32);
      if (ov < v || (ov == v && oi < i)) { v = ov; i = oi; }
    }
    bestv[r] = v; besti[r] = i;
  }
  if (nl == 0) {
#pragma unroll
    for (int r = 0; r < 8; ++r) {
      const int row = rowbase + half * 8 + r;   // half0: M=r, half1: M=8+r
      idx_i[row] = besti[r];
      idx_f[row] = (float)besti[r];
    }
  }
}

// ---------------------------------------------------------------------------
// Kernel 3: gather z_q (straight-through output == z_q), commitment loss,
// counts and dw scatter-adds. Threads map to consecutive hw -> coalesced.
// ---------------------------------------------------------------------------
__global__ void __launch_bounds__(256)
assign_kernel(const float* __restrict__ z, const float* __restrict__ emb,
              const int* __restrict__ idx_i, float* __restrict__ zq_out,
              float* __restrict__ counts, float* __restrict__ dw,
              float* __restrict__ loss) {
  const int n  = blockIdx.x * 256 + threadIdx.x;
  const int b  = n >> 10;
  const int hw = n & (HW - 1);
  const int k  = idx_i[n];
  atomicAdd(&counts[k], 1.0f);

  const size_t zoff = (size_t)b * DIM * HW + hw;
  float lsum = 0.0f;
  for (int c = 0; c < DIM; ++c) {
    float zv = z[zoff + (size_t)c * HW];
    float ev = emb[k * DIM + c];
    zq_out[zoff + (size_t)c * HW] = ev;          // z_q_st == z_q numerically
    float d = ev - zv;
    lsum += d * d;
    atomicAdd(&dw[k * DIM + c], zv);             // segment_sum(zf, idx)
  }

  __shared__ float red[256];
  red[threadIdx.x] = lsum;
  __syncthreads();
  for (int s = 128; s > 0; s >>= 1) {
    if (threadIdx.x < s) red[threadIdx.x] += red[threadIdx.x + s];
    __syncthreads();
  }
  if (threadIdx.x == 0)
    atomicAdd(loss, red[0] * (BETA / ((float)NVEC * (float)DIM)));
}

// ---------------------------------------------------------------------------
// Kernel 4: EMA cluster sizes + Laplace smoothing (single block over K)
// ---------------------------------------------------------------------------
__global__ void __launch_bounds__(KCODE)
cs_kernel(const float* __restrict__ ema_cs, const float* __restrict__ counts,
          float* __restrict__ cs_out) {
  const int k = threadIdx.x;
  float v = ema_cs[k] * DECAY + (1.0f - DECAY) * counts[k];
  __shared__ float red[KCODE];
  red[k] = v;
  __syncthreads();
  for (int s = KCODE / 2; s > 0; s >>= 1) {
    if (k < s) red[k] += red[k + s];
    __syncthreads();
  }
  const float nsum = red[0];
  cs_out[k] = (v + EPSV) / (nsum + (float)KCODE * EPSV) * nsum;
}

// ---------------------------------------------------------------------------
// Kernel 5: new_ema_w and new_embedding
// ---------------------------------------------------------------------------
__global__ void __launch_bounds__(256)
finalize_kernel(const float* __restrict__ ema_w, const float* __restrict__ dw,
                const float* __restrict__ cs, float* __restrict__ emaw_out,
                float* __restrict__ emb_out) {
  const int i = blockIdx.x * blockDim.x + threadIdx.x;  // [0, K*D)
  const int k = i >> 8;
  float w = ema_w[i] * DECAY + (1.0f - DECAY) * dw[i];
  emaw_out[i] = w;
  emb_out[i]  = w / cs[k];
}

// ---------------------------------------------------------------------------
extern "C" void kernel_launch(void* const* d_in, const int* in_sizes, int n_in,
                              void* d_out, int out_size, void* d_ws, size_t ws_size,
                              hipStream_t stream) {
  const float* z      = (const float*)d_in[0];  // [32,256,32,32]
  const float* emb    = (const float*)d_in[1];  // [1024,256]
  const float* ema_cs = (const float*)d_in[2];  // [1024]
  const float* ema_w  = (const float*)d_in[3];  // [1024,256]

  // Outputs (float32, concatenated in reference return order)
  float* out    = (float*)d_out;
  float* o_zq   = out;                 // 8,388,608
  float* o_idx  = out + 8388608;       //    32,768
  float* o_loss = out + 8421376;       //         1
  float* o_emb  = out + 8421377;       //   262,144
  float* o_cs   = out + 8683521;       //     1,024
  float* o_emaw = out + 8684545;       //   262,144

  // Workspace layout
  char*   ws     = (char*)d_ws;
  __bf16* ebf    = (__bf16*)ws;                 // 512 KB  bf16 codebook
  float*  enorm  = (float*)(ws + 524288);       //   4 KB  ||e||^2
  int*    idx_i  = (int*)  (ws + 528384);       // 128 KB  argmin indices
  float*  counts = (float*)(ws + 659456);       //   4 KB
  float*  dw     = (float*)(ws + 663552);       //   1 MB  segment sums

  prep_codebook<<<KCODE, DIM, 0, stream>>>(emb, ebf, enorm);
  zero_ws<<<(KCODE * DIM + 255) / 256, 256, 0, stream>>>(counts, dw, o_loss);
  argmin_kernel<<<NVEC / 128, 256, 0, stream>>>(z, ebf, enorm, idx_i, o_idx);
  assign_kernel<<<NVEC / 256, 256, 0, stream>>>(z, emb, idx_i, o_zq, counts, dw, o_loss);
  cs_kernel<<<1, KCODE, 0, stream>>>(ema_cs, counts, o_cs);
  finalize_kernel<<<KCODE * DIM / 256, 256, 0, stream>>>(ema_w, dw, o_cs, o_emaw, o_emb);
}
